// MultiHeadAttention_72825465471225
// MI455X (gfx1250) — matrix operand order repaired and checked by reading, not compiled, not run on hardware
//
#include <hip/hip_runtime.h>

// ---------- types ----------
typedef __bf16 bf16;
typedef __attribute__((ext_vector_type(16))) __bf16 bf16x16;
typedef __attribute__((ext_vector_type(8)))  __bf16 bf16x8;
typedef __attribute__((ext_vector_type(8)))  float  f32x8;
typedef int v4i __attribute__((vector_size(16)));   // b128 payload type expected by async builtin

__device__ __forceinline__ bf16x16 combine16(bf16x8 lo, bf16x8 hi) {
  return __builtin_shufflevector(lo, hi, 0,1,2,3,4,5,6,7,8,9,10,11,12,13,14,15);
}

// xor-butterfly via ds_swizzle (group-of-32: out = ((lane & 0x1f) | 0) ^ m)
#define SWZ_XOR(x, m) \
  __int_as_float(__builtin_amdgcn_ds_swizzle(__float_as_int(x), (((m) << 10) | 0x1f)))

// ---------- CDNA5 async global->LDS path (guarded; sync fallback) ----------
#if defined(__gfx1250__) && __has_builtin(__builtin_amdgcn_global_load_async_to_lds_b128)
#define ASYNC_LDS 1
#endif

#if defined(ASYNC_LDS)
__device__ __forceinline__ void async_copy16(const bf16* g, bf16* l) {
  __builtin_amdgcn_global_load_async_to_lds_b128(
      (__attribute__((address_space(1))) v4i*)(g),
      (__attribute__((address_space(3))) v4i*)(l), 0, 0);
}
__device__ __forceinline__ void wait_async3() {
#if __has_builtin(__builtin_amdgcn_s_wait_asynccnt)
  __builtin_amdgcn_s_wait_asynccnt(3);
#else
  asm volatile("s_wait_asynccnt 0x3" ::: "memory");
#endif
}
__device__ __forceinline__ void wait_async0() {
#if __has_builtin(__builtin_amdgcn_s_wait_asynccnt)
  __builtin_amdgcn_s_wait_asynccnt(0);
#else
  asm volatile("s_wait_asynccnt 0x0" ::: "memory");
#endif
}
#endif

// ---------- fp32 -> bf16 converter ----------
__global__ void cvt_f32_bf16(const float* __restrict__ in, bf16* __restrict__ out, int n) {
  int i = blockIdx.x * blockDim.x + threadIdx.x;
  int stride = gridDim.x * blockDim.x;
  for (; i < n; i += stride) out[i] = (bf16)in[i];
}

// ---------- GEMM: Y = (X @ W^T + bias) * outScale ----------
// X: [M][K] bf16, W: [Nout][K] bf16, bias: [Nout] f32.
// Block tile 64(M) x 128(N), 8 waves (2x4), each wave 32x32 via 2x2 WMMA accums.
#define BK   32
#define LSTR 40   // padded LDS row stride (bf16 elems); 80B keeps 16B alignment

__device__ __forceinline__ void mm_step(const bf16* __restrict__ Xs,
                                        const bf16* __restrict__ Ws,
                                        int wm, int wn, int r, int hf,
                                        f32x8 (&acc)[2][2]) {
  bf16x16 a[2], b[2];
#pragma unroll
  for (int i = 0; i < 2; ++i) {
    int row = wm * 32 + i * 16 + r;
    bf16x8 lo = *(const bf16x8*)(&Xs[row * LSTR + hf * 8]);
    bf16x8 hi = *(const bf16x8*)(&Xs[row * LSTR + 16 + hf * 8]);
    a[i] = combine16(lo, hi);
  }
#pragma unroll
  for (int j = 0; j < 2; ++j) {
    int col = wn * 32 + j * 16 + r;  // B column c = W row (K-contiguous)
    bf16x8 lo = *(const bf16x8*)(&Ws[col * LSTR + hf * 8]);
    bf16x8 hi = *(const bf16x8*)(&Ws[col * LSTR + 16 + hf * 8]);
    b[j] = combine16(lo, hi);
  }
#pragma unroll
  for (int i = 0; i < 2; ++i)
#pragma unroll
    for (int j = 0; j < 2; ++j)
      acc[i][j] = __builtin_amdgcn_wmma_f32_16x16x32_bf16(
          false, a[i], false, b[j], (short)0, acc[i][j], false, false);
}

template <bool OUT_F32>
__global__ __launch_bounds__(256) void gemm_xwt(
    const bf16* __restrict__ X, const bf16* __restrict__ W,
    const float* __restrict__ bias,
    float* __restrict__ Yf, bf16* __restrict__ Yb,
    int M, int Nout, int K, float outScale)
{
  const int tid  = threadIdx.x;
  const int lane = tid & 31;
  const int wave = tid >> 5;
  const int wm   = wave & 1;   // 0..1 (M)
  const int wn   = wave >> 1;  // 0..3 (N)
  const int m0   = blockIdx.x * 64;
  const int n0   = blockIdx.y * 128;
  const int r    = lane & 15;
  const int hf   = lane >> 4;
  const int xr   = tid >> 2;          // X tile: 64x32 = 256 chunks of 8
  const int xc   = (tid & 3) * 8;

  f32x8 acc[2][2] = {};

#if defined(ASYNC_LDS)
  // double-buffered async staging: copy tile i+1 while WMMA consumes tile i
  __shared__ __align__(16) bf16 Xs[2][64 * LSTR];
  __shared__ __align__(16) bf16 Ws[2][128 * LSTR];

  auto issue = [&](int k0, int p) {
    async_copy16(X + (size_t)(m0 + xr) * K + k0 + xc, &Xs[p][xr * LSTR + xc]);
#pragma unroll
    for (int c = 0; c < 2; ++c) {
      int ch = tid + c * 256;                   // W tile: 128x32 = 512 chunks
      int row = ch >> 2, col = (ch & 3) * 8;
      async_copy16(W + (size_t)(n0 + row) * K + k0 + col, &Ws[p][row * LSTR + col]);
    }
  };
  const int nk = K / BK;
  issue(0, 0);
  for (int i = 0; i < nk; ++i) {
    int p = i & 1;
    if (i + 1 < nk) { issue((i + 1) * BK, p ^ 1); wait_async3(); }
    else            { wait_async0(); }
    __syncthreads();                 // all waves' tile-i copies landed
    mm_step(Xs[p], Ws[p], wm, wn, r, hf, acc);
    __syncthreads();                 // safe to overwrite buffer p next round
  }
#else
  __shared__ __align__(16) bf16 Xs[64 * LSTR];
  __shared__ __align__(16) bf16 Ws[128 * LSTR];
  for (int k0 = 0; k0 < K; k0 += BK) {
    *(bf16x8*)(&Xs[xr * LSTR + xc]) =
        *(const bf16x8*)(X + (size_t)(m0 + xr) * K + k0 + xc);
#pragma unroll
    for (int c = 0; c < 2; ++c) {
      int ch = tid + c * 256;
      int row = ch >> 2, col = (ch & 3) * 8;
      *(bf16x8*)(&Ws[row * LSTR + col]) =
          *(const bf16x8*)(W + (size_t)(n0 + row) * K + k0 + col);
    }
    if (k0 + BK < K) {
      __builtin_prefetch(X + (size_t)(m0 + xr) * K + k0 + BK + xc, 0, 1);
      __builtin_prefetch(W + (size_t)(n0 + (tid >> 1)) * K + k0 + BK, 0, 1);
    }
    __syncthreads();
    mm_step(Xs, Ws, wm, wn, r, hf, acc);
    __syncthreads();
  }
#endif

  // epilogue: C layout -> lane holds (m = v + 8*hf, n = r); lanes of a half
  // cover 16 consecutive columns of one row -> stores coalesce across lanes.
#pragma unroll
  for (int i = 0; i < 2; ++i)
#pragma unroll
    for (int j = 0; j < 2; ++j) {
      int n = n0 + wn * 32 + j * 16 + r;
      float bv = bias[n];
#pragma unroll
      for (int v = 0; v < 8; ++v) {
        int m = m0 + wm * 32 + i * 16 + v + 8 * hf;
        float val = (acc[i][j][v] + bv) * outScale;
        if constexpr (OUT_F32) Yf[(size_t)m * Nout + n] = val;
        else                   Yb[(size_t)m * Nout + n] = (bf16)val;
      }
    }
}

// ---------- Flash attention per head ----------
// Q,K,V: [N][1024] bf16 (head h = cols h*64..h*64+63); Q pre-scaled by 1/8.
// Block: 128 query rows of one head; 8 waves x 16 rows. Key loop in tiles of 64.
__global__ __launch_bounds__(256) void mha_attn(
    const bf16* __restrict__ Q, const bf16* __restrict__ K,
    const bf16* __restrict__ V, bf16* __restrict__ O, int N, int Dm)
{
  const int h    = blockIdx.y;
  const int q0   = blockIdx.x * 128;
  const int tid  = threadIdx.x;
  const int lane = tid & 31;
  const int wave = tid >> 5;
  const int r    = lane & 15;
  const int hf   = lane >> 4;
  const int co   = h * 64;

  __shared__ __align__(16) bf16 Ks[64 * 72];      // [key][d]   row-major
  __shared__ __align__(16) bf16 Vt[64 * 72];      // [d][key]   transposed
  __shared__ __align__(16) bf16 Ps[8][16 * 64];   // per-wave P tile

  // Q fragments (already scaled by 1/sqrt(d_k) in the projection epilogue)
  bf16x16 qa[2];
  {
    const bf16* qp = Q + (size_t)(q0 + wave * 16 + r) * Dm + co;
#pragma unroll
    for (int kd = 0; kd < 2; ++kd) {
      bf16x8 lo = *(const bf16x8*)(qp + kd * 32 + hf * 8);
      bf16x8 hi = *(const bf16x8*)(qp + kd * 32 + 16 + hf * 8);
      qa[kd] = combine16(lo, hi);
    }
  }

  // all-ones B fragment: l = P @ 1 via WMMA (row-sum replicated to all lanes)
  bf16x16 ones;
#pragma unroll
  for (int e = 0; e < 16; ++e) ones[e] = (bf16)1.0f;

  f32x8 o_acc[4] = {};
  f32x8 l_acc = {};
  f32x8 m_i;
#pragma unroll
  for (int v = 0; v < 8; ++v) m_i[v] = -3.0e38f;

  for (int kb = 0; kb < N; kb += 64) {
    // stage K block (async direct-to-LDS when available) + V transposed
#pragma unroll
    for (int c = 0; c < 2; ++c) {
      int ch = tid + c * 256;
      int row = ch >> 2, col = (ch & 3) * 8;
#if defined(ASYNC_LDS)
      async_copy16(K + (size_t)(kb + row) * Dm + co + col, &Ks[row * 72 + col]);
#else
      *(bf16x8*)(&Ks[row * 72 + col]) =
          *(const bf16x8*)(K + (size_t)(kb + row) * Dm + co + col);
#endif
      bf16x8 vv = *(const bf16x8*)(V + (size_t)(kb + row) * Dm + co + col);
#pragma unroll
      for (int e = 0; e < 8; ++e) Vt[(col + e) * 72 + row] = vv[e];
    }
#if defined(ASYNC_LDS)
    wait_async0();
#endif
    __syncthreads();

    // S = Q K^T : 16x64 per wave (B column c = key row, d-contiguous)
    f32x8 s[4] = {};
#pragma unroll
    for (int kd = 0; kd < 2; ++kd)
#pragma unroll
      for (int j = 0; j < 4; ++j) {
        int col = j * 16 + r;
        bf16x8 lo = *(const bf16x8*)(&Ks[col * 72 + kd * 32 + hf * 8]);
        bf16x8 hi = *(const bf16x8*)(&Ks[col * 72 + kd * 32 + 16 + hf * 8]);
        bf16x16 b = combine16(lo, hi);
        s[j] = __builtin_amdgcn_wmma_f32_16x16x32_bf16(
            false, qa[kd], false, b, (short)0, s[j], false, false);
      }

    // row max: stage-major xor butterfly over 16 lanes -> 8 independent
    // swizzles per stage (one dscnt wait per stage instead of per element)
    float t[8], u[8];
#pragma unroll
    for (int v = 0; v < 8; ++v)
      t[v] = fmaxf(fmaxf(s[0][v], s[1][v]), fmaxf(s[2][v], s[3][v]));
#pragma unroll
    for (int v = 0; v < 8; ++v) u[v] = SWZ_XOR(t[v], 1);
#pragma unroll
    for (int v = 0; v < 8; ++v) t[v] = fmaxf(t[v], u[v]);
#pragma unroll
    for (int v = 0; v < 8; ++v) u[v] = SWZ_XOR(t[v], 2);
#pragma unroll
    for (int v = 0; v < 8; ++v) t[v] = fmaxf(t[v], u[v]);
#pragma unroll
    for (int v = 0; v < 8; ++v) u[v] = SWZ_XOR(t[v], 4);
#pragma unroll
    for (int v = 0; v < 8; ++v) t[v] = fmaxf(t[v], u[v]);
#pragma unroll
    for (int v = 0; v < 8; ++v) u[v] = SWZ_XOR(t[v], 8);
#pragma unroll
    for (int v = 0; v < 8; ++v) t[v] = fmaxf(t[v], u[v]);

    f32x8 corr;
#pragma unroll
    for (int v = 0; v < 8; ++v) {
      float mn = fmaxf(m_i[v], t[v]);
      corr[v]  = __expf(m_i[v] - mn);
      m_i[v]   = mn;
    }
#pragma unroll
    for (int j = 0; j < 4; ++j)
#pragma unroll
      for (int v = 0; v < 8; ++v) o_acc[j][v] *= corr[v];
#pragma unroll
    for (int v = 0; v < 8; ++v) l_acc[v] *= corr[v];

    // P = exp(S - m): C-layout f32 -> bf16 A-layout via per-wave LDS tile
#pragma unroll
    for (int j = 0; j < 4; ++j)
#pragma unroll
      for (int v = 0; v < 8; ++v)
        Ps[wave][(v + 8 * hf) * 64 + j * 16 + r] = (bf16)__expf(s[j][v] - m_i[v]);

    // O += P @ V ; l += P @ 1   (B column c key-contiguous in Vt)
#pragma unroll
    for (int kd = 0; kd < 2; ++kd) {
      bf16x8 lo = *(const bf16x8*)(&Ps[wave][r * 64 + kd * 32 + hf * 8]);
      bf16x8 hi = *(const bf16x8*)(&Ps[wave][r * 64 + kd * 32 + 16 + hf * 8]);
      bf16x16 pa = combine16(lo, hi);
#pragma unroll
      for (int j = 0; j < 4; ++j) {
        int col = j * 16 + r;
        bf16x8 blo = *(const bf16x8*)(&Vt[col * 72 + kd * 32 + hf * 8]);
        bf16x8 bhi = *(const bf16x8*)(&Vt[col * 72 + kd * 32 + 16 + hf * 8]);
        bf16x16 b = combine16(blo, bhi);
        o_acc[j] = __builtin_amdgcn_wmma_f32_16x16x32_bf16(
            false, pa, false, b, (short)0, o_acc[j], false, false);
      }
      l_acc = __builtin_amdgcn_wmma_f32_16x16x32_bf16(
          false, pa, false, ones, (short)0, l_acc, false, false);
    }
    __syncthreads();
  }

  // normalize and write concat buffer (stores coalesce across the 16 lanes)
#pragma unroll
  for (int j = 0; j < 4; ++j)
#pragma unroll
    for (int v = 0; v < 8; ++v) {
      int m = q0 + wave * 16 + v + 8 * hf;
      int c = co + j * 16 + r;
      O[(size_t)m * Dm + c] = (bf16)(o_acc[j][v] / l_acc[v]);
    }
}

// ---------- launcher ----------
extern "C" void kernel_launch(void* const* d_in, const int* in_sizes, int n_in,
                              void* d_out, int out_size, void* d_ws, size_t ws_size,
                              hipStream_t stream)
{
  (void)in_sizes; (void)n_in; (void)out_size; (void)ws_size;
  const float* q  = (const float*)d_in[0];
  const float* k  = (const float*)d_in[1];
  const float* v  = (const float*)d_in[2];
  const float* Wq = (const float*)d_in[3];
  const float* bq = (const float*)d_in[4];
  const float* Wk = (const float*)d_in[5];
  const float* bk = (const float*)d_in[6];
  const float* Wv = (const float*)d_in[7];
  const float* bv = (const float*)d_in[8];
  const float* Wo = (const float*)d_in[9];
  const float* bo = (const float*)d_in[10];
  float* out = (float*)d_out;

  const int N = 2048, D = 1024;
  size_t off = 0;
  auto alloc = [&](size_t elems) {
    bf16* p = (bf16*)((char*)d_ws + off);
    off += elems * sizeof(bf16);
    return p;
  };
  bf16* qb  = alloc((size_t)N * D);
  bf16* kb  = alloc((size_t)N * D);
  bf16* vb  = alloc((size_t)N * D);
  bf16* Wqb = alloc((size_t)D * D);
  bf16* Wkb = alloc((size_t)D * D);
  bf16* Wvb = alloc((size_t)D * D);
  bf16* Wob = alloc((size_t)D * D);
  bf16* Qb  = alloc((size_t)N * D);
  bf16* Kb  = alloc((size_t)N * D);
  bf16* Vb  = alloc((size_t)N * D);
  bf16* Cb  = alloc((size_t)N * D);

  cvt_f32_bf16<<<1024, 256, 0, stream>>>(q,  qb,  N * D);
  cvt_f32_bf16<<<1024, 256, 0, stream>>>(k,  kb,  N * D);
  cvt_f32_bf16<<<1024, 256, 0, stream>>>(v,  vb,  N * D);
  cvt_f32_bf16<<<1024, 256, 0, stream>>>(Wq, Wqb, D * D);
  cvt_f32_bf16<<<1024, 256, 0, stream>>>(Wk, Wkb, D * D);
  cvt_f32_bf16<<<1024, 256, 0, stream>>>(Wv, Wvb, D * D);
  cvt_f32_bf16<<<1024, 256, 0, stream>>>(Wo, Wob, D * D);

  dim3 gb(N / 64, D / 128);  // 32 x 8
  // Q projection folds the 1/sqrt(d_k)=0.125 attention scale into its output.
  gemm_xwt<false><<<gb, 256, 0, stream>>>(qb, Wqb, bq, nullptr, Qb, N, D, D, 0.125f);
  gemm_xwt<false><<<gb, 256, 0, stream>>>(kb, Wkb, bk, nullptr, Kb, N, D, D, 1.0f);
  gemm_xwt<false><<<gb, 256, 0, stream>>>(vb, Wvb, bv, nullptr, Vb, N, D, D, 1.0f);

  dim3 ga(N / 128, 16);      // 16 query blocks x 16 heads
  mha_attn<<<ga, 256, 0, stream>>>(Qb, Kb, Vb, Cb, N, D);

  gemm_xwt<true><<<gb, 256, 0, stream>>>(Cb, Wob, bo, out, nullptr, N, D, D, 1.0f);
}
